// PAM_4320737100642
// MI455X (gfx1250) — matrix operand order repaired and checked by reading, not compiled, not run on hardware
//
#include <hip/hip_runtime.h>

// PAM attention for MI455X (gfx1250, wave32, WMMA bf16).
// B=4, C=512, H=W=64 -> N=4096, DQK=64.
// f32 accumulate everywhere, bf16 on WMMA operands only.

#define PB 4
#define PC 512
#define PN 4096
#define PD 64

typedef __attribute__((ext_vector_type(16))) __bf16 v16bf;
typedef __attribute__((ext_vector_type(8)))  float  v8f;

#define WMMA_BF16(A, Bf, Cacc) \
    __builtin_amdgcn_wmma_f32_16x16x32_bf16(false, (A), false, (Bf), (short)0, (Cacc), false, false)

// --------------------------- async copy to LDS ------------------------------
#if defined(__AMDGCN__) && __has_builtin(__builtin_amdgcn_global_load_async_to_lds_b128)
#define HAVE_ASYNC_LDS 1
#else
#define HAVE_ASYNC_LDS 0
#endif

typedef int b128_t __attribute__((vector_size(16)));
typedef __attribute__((address_space(1))) b128_t gb128_t;   // global int4
typedef __attribute__((address_space(3))) b128_t lb128_t;   // LDS int4

__device__ __forceinline__ void cp16_to_lds(void* lds_dst, const void* gsrc) {
#if HAVE_ASYNC_LDS
    gb128_t* g = (gb128_t*)(unsigned long long)gsrc;
    lb128_t* l = (lb128_t*)(unsigned int)(unsigned long long)lds_dst;
    __builtin_amdgcn_global_load_async_to_lds_b128(g, l, 0, 0);
#else
    *(float4*)lds_dst = *(const float4*)gsrc;
#endif
}

__device__ __forceinline__ void wait_async_lds() {
#if HAVE_ASYNC_LDS
    asm volatile("s_wait_asynccnt 0x0" ::: "memory");
#endif
}

// --------------------------- fragment loaders -------------------------------
// 16 contiguous bf16 starting at p (B-fragment rows / contiguous K slices)
__device__ __forceinline__ v16bf frag_contig(const __bf16* p) {
    union { uint4 u[2]; v16bf v; } f;
    f.u[0] = *(const uint4*)p;
    f.u[1] = *(const uint4*)(p + 8);
    return f.v;
}
// two 8-element contiguous chunks (A-fragment per-lane K pattern)
__device__ __forceinline__ v16bf frag_pair(const __bf16* p0, const __bf16* p1) {
    union { uint4 u[2]; v16bf v; } f;
    f.u[0] = *(const uint4*)p0;
    f.u[1] = *(const uint4*)p1;
    return f.v;
}

// ---------------------------------------------------------------------------
// Kernel 1: fused QKV projection.
//   Q[n][d], K[n][d] bf16 (row per position -> contiguous fragment loads),
//   V[c][n]  bf16 (natural layout -> contiguous O-GEMM B-fragments).
//   Grid: (N/64, (64+64+512)/16 = 40, B), block 128 (4 waves).
//   x/W tiles staged with global_load_async_to_lds_b128.
// ---------------------------------------------------------------------------
__global__ __launch_bounds__(128) void pam_proj(
    const float* __restrict__ x,
    const float* __restrict__ Wq, const float* __restrict__ bq,
    const float* __restrict__ Wk, const float* __restrict__ bk,
    const float* __restrict__ Wv, const float* __restrict__ bv,
    __bf16* __restrict__ Q, __bf16* __restrict__ K, __bf16* __restrict__ V2)
{
    __shared__ float lw[16][36];   // [m][c] W tile, rows 144B (16B aligned)
    __shared__ float lx[32][68];   // [c][n] x tile, rows 272B (16B aligned)

    const int b  = blockIdx.z;
    const int n0 = blockIdx.x * 64;
    const int mt = blockIdx.y * 16;

    const float* W;
    const float* bias;
    __bf16* dst;
    int mbase, isV;
    if (mt < 64)       { W = Wq; bias = bq; dst = Q;  mbase = mt;       isV = 0; }
    else if (mt < 128) { W = Wk; bias = bk; dst = K;  mbase = mt - 64;  isV = 0; }
    else               { W = Wv; bias = bv; dst = V2; mbase = mt - 128; isV = 1; }

    const int t    = threadIdx.x;
    const int wave = t >> 5;
    const int lane = t & 31;
    const int half = lane >> 4;
    const int l    = lane & 15;

    const float* xb = x + (size_t)b * PC * PN;

    v8f acc = {};
    for (int c0 = 0; c0 < PC; c0 += 32) {
        // stage W tile (16x32 f32): one 16B chunk per thread
        {
            const int m = t >> 3, coff = (t & 7) * 4;
            cp16_to_lds(&lw[m][coff], &W[(size_t)(mbase + m) * PC + c0 + coff]);
        }
        // stage x tile (32x64 f32): four 16B chunks per thread
#pragma unroll
        for (int it = 0; it < 4; ++it) {
            const int idx = t + it * 128;
            const int r = idx >> 4, coff = (idx & 15) * 4;
            cp16_to_lds(&lx[r][coff], xb + (size_t)(c0 + r) * PN + n0 + coff);
        }
        wait_async_lds();
        __syncthreads();

        // A fragment: rows = positions, K = channels (column reads + cvt)
        v16bf a, w;
#pragma unroll
        for (int e = 0; e < 16; ++e) {
            const int k = e + (e & 8) + 8 * half;   // {0..7,16..23}/{8..15,24..31}
            a[e] = (__bf16)lx[k][wave * 16 + l];
        }
        // B fragment: K = channels, cols = output channels
#pragma unroll
        for (int e = 0; e < 16; ++e)
            w[e] = (__bf16)lw[l][e + (e & 8) + 8 * half];

        acc = WMMA_BF16(a, w, acc);
        __syncthreads();
    }

    const float bb = bias[mbase + l];
    if (!isV) {
        // Q/K: [n][64], lane owns column d = mbase+l
#pragma unroll
        for (int r = 0; r < 8; ++r) {
            const int i = n0 + wave * 16 + r + 8 * half;
            dst[((size_t)b * PN + i) * PD + mbase + l] = (__bf16)(acc[r] + bb);
        }
    } else {
        // V: [c][n], lane owns channel c = mbase+l, 8 consecutive n -> one b128
        union { unsigned short s[8]; uint4 u; } pk;
#pragma unroll
        for (int r = 0; r < 8; ++r) {
            const __bf16 h = (__bf16)(acc[r] + bb);
            pk.s[r] = __builtin_bit_cast(unsigned short, h);
        }
        __bf16* p = dst + ((size_t)b * PC + mbase + l) * PN + n0 + wave * 16 + 8 * half;
        *(uint4*)p = pk.u;
    }
}

// ---------------------------------------------------------------------------
// Kernel 2: softmax statistics (row max and row sum of exp(S-m)).
//   S^T tiles (A = K_j, B = Q_i^T) -> lane-local online softmax, halves
//   combined with shfl_xor(16). Grid: (N/128, B), block 256.
// ---------------------------------------------------------------------------
__global__ __launch_bounds__(256) void pam_stats(
    const __bf16* __restrict__ Q, const __bf16* __restrict__ K,
    float* __restrict__ stats_m, float* __restrict__ stats_s)
{
    const int b    = blockIdx.y;
    const int t    = threadIdx.x;
    const int wave = t >> 5;
    const int lane = t & 31;
    const int half = lane >> 4;
    const int l    = lane & 15;
    const int i0   = blockIdx.x * 128 + wave * 16;

    const __bf16* qrow = Q + ((size_t)b * PN + i0 + l) * PD;
    const v16bf q0 = frag_pair(qrow + 8 * half, qrow + 16 + 8 * half);        // B-frag, d-chunk 0
    const v16bf q1 = frag_pair(qrow + 32 + 8 * half, qrow + 48 + 8 * half);   // B-frag, d-chunk 1

    float m_run = -1e30f, s_run = 0.0f;
    for (int j0 = 0; j0 < PN; j0 += 16) {
        const __bf16* krow = K + ((size_t)b * PN + j0 + l) * PD;
        const v16bf k0 = frag_pair(krow + 8 * half, krow + 16 + 8 * half);
        const v16bf k1 = frag_pair(krow + 32 + 8 * half, krow + 48 + 8 * half);
        v8f s = {};
        s = WMMA_BF16(k0, q0, s);
        s = WMMA_BF16(k1, q1, s);

        float tm = m_run;
#pragma unroll
        for (int r = 0; r < 8; ++r) tm = fmaxf(tm, s[r]);
        float ts = s_run * __expf(m_run - tm);
#pragma unroll
        for (int r = 0; r < 8; ++r) ts += __expf(s[r] - tm);
        m_run = tm;
        s_run = ts;
    }

    const float m_o = __shfl_xor(m_run, 16, 32);
    const float s_o = __shfl_xor(s_run, 16, 32);
    const float mt  = fmaxf(m_run, m_o);
    const float st  = s_run * __expf(m_run - mt) + s_o * __expf(m_o - mt);
    if (half == 0) {
        stats_m[(size_t)b * PN + i0 + l] = mt;
        stats_s[(size_t)b * PN + i0 + l] = st;
    }
}

// ---------------------------------------------------------------------------
// Kernel 3: O = P @ V^T, epilogue gamma*O + x.
//   Workgroup: 8 waves = 2 i-tiles x 4 c-blocks(128ch). Per 128-key strip the
//   4 waves sharing an i-tile each produce a distinct 32-key slice of the
//   shared P strip in LDS (P computed exactly once -> S overhead 12.5%).
//   O-GEMM B-fragments load straight from V[c][n] (contiguous 32B per lane,
//   L2-resident). Grid: (N/32, B), block 256.
// ---------------------------------------------------------------------------
__global__ __launch_bounds__(256) void pam_out(
    const float* __restrict__ x,
    const __bf16* __restrict__ Q, const __bf16* __restrict__ K,
    const __bf16* __restrict__ V2,
    const float* __restrict__ stats_m, const float* __restrict__ stats_s,
    const float* __restrict__ gamma,
    float* __restrict__ out)
{
    __shared__ __bf16 lp[2][16][136];   // P strips [i-tile][i][128 keys], 272B rows

    const int b    = blockIdx.y;
    const int t    = threadIdx.x;
    const int wave = t >> 5;
    const int lane = t & 31;
    const int half = lane >> 4;
    const int l    = lane & 15;
    const int iw   = wave >> 2;          // i-tile within workgroup (0..1)
    const int cw   = wave & 3;           // c-block (0..3) and P-producer slice
    const int i0   = blockIdx.x * 32 + iw * 16;
    const int cb0  = cw * 128;

    const float g = gamma[0];

    float mrow[8], sinv[8];
#pragma unroll
    for (int r = 0; r < 8; ++r) {
        const int i = i0 + r + 8 * half;
        mrow[r] = stats_m[(size_t)b * PN + i];
        sinv[r] = 1.0f / stats_s[(size_t)b * PN + i];
    }

    const __bf16* qrow = Q + ((size_t)b * PN + i0 + l) * PD;
    const v16bf qa0 = frag_contig(qrow + 16 * half);
    const v16bf qa1 = frag_contig(qrow + 32 + 16 * half);

    v8f o[8];
    {
        const v8f z = {};
#pragma unroll
        for (int ct = 0; ct < 8; ++ct) o[ct] = z;
    }

    for (int js = 0; js < PN; js += 128) {
        // ---- producer: this wave's 32-key slice of the P strip ----
        const int jp = js + cw * 32;
#pragma unroll
        for (int jt = 0; jt < 2; ++jt) {
            const __bf16* krow = K + ((size_t)b * PN + jp + jt * 16 + l) * PD;
            const v16bf kb0 = frag_contig(krow + 16 * half);
            const v16bf kb1 = frag_contig(krow + 32 + 16 * half);
            v8f s = {};
            s = WMMA_BF16(qa0, kb0, s);
            s = WMMA_BF16(qa1, kb1, s);
#pragma unroll
            for (int r = 0; r < 8; ++r) {
                const float p = __expf(s[r] - mrow[r]) * sinv[r];
                lp[iw][r + 8 * half][cw * 32 + jt * 16 + l] = (__bf16)p;
            }
        }
        __syncthreads();

        // ---- consumer: O += P(16x32) @ V^T(32x16) over this wave's 128 ch ----
#pragma unroll
        for (int jc = 0; jc < 4; ++jc) {
            const v16bf pa = frag_contig(&lp[iw][l][jc * 32 + 16 * half]);
#pragma unroll
            for (int ct = 0; ct < 8; ++ct) {
                const __bf16* vrow = V2 + ((size_t)b * PC + cb0 + ct * 16 + l) * PN
                                        + js + jc * 32 + 16 * half;
                const v16bf vb = frag_contig(vrow);
                o[ct] = WMMA_BF16(pa, vb, o[ct]);
            }
        }
        __syncthreads();
    }

    // epilogue: lane owns channel c -> 8 consecutive n per lane = 2 float4
#pragma unroll
    for (int ct = 0; ct < 8; ++ct) {
        const int c = cb0 + ct * 16 + l;
        const size_t base = ((size_t)b * PC + c) * PN + i0 + 8 * half;
        const float4 x0 = *(const float4*)(x + base);
        const float4 x1 = *(const float4*)(x + base + 4);
        float4 r0, r1;
        r0.x = g * o[ct][0] + x0.x;  r0.y = g * o[ct][1] + x0.y;
        r0.z = g * o[ct][2] + x0.z;  r0.w = g * o[ct][3] + x0.w;
        r1.x = g * o[ct][4] + x1.x;  r1.y = g * o[ct][5] + x1.y;
        r1.z = g * o[ct][6] + x1.z;  r1.w = g * o[ct][7] + x1.w;
        *(float4*)(out + base)     = r0;
        *(float4*)(out + base + 4) = r1;
    }
}

// ---------------------------------------------------------------------------
extern "C" void kernel_launch(void* const* d_in, const int* in_sizes, int n_in,
                              void* d_out, int out_size, void* d_ws, size_t ws_size,
                              hipStream_t stream)
{
    (void)in_sizes; (void)n_in; (void)out_size; (void)ws_size;

    const float* x     = (const float*)d_in[0];
    const float* Wq    = (const float*)d_in[1];
    const float* bq    = (const float*)d_in[2];
    const float* Wk    = (const float*)d_in[3];
    const float* bk    = (const float*)d_in[4];
    const float* Wv    = (const float*)d_in[5];
    const float* bv    = (const float*)d_in[6];
    const float* gamma = (const float*)d_in[7];
    float* out = (float*)d_out;

    // workspace layout (bytes):
    //   stats_m : B*N f32          (  64 KB)
    //   stats_s : B*N f32          (  64 KB)
    //   Q       : B*N*64   bf16    (   2 MB)
    //   K       : B*N*64   bf16    (   2 MB)
    //   V2      : B*C*N    bf16    (  16 MB)   -- channel-major [b][c][n]
    char* ws = (char*)d_ws;
    float*  stats_m = (float*)ws;
    float*  stats_s = (float*)(ws + (size_t)PB * PN * 4);
    __bf16* Q  = (__bf16*)(ws + (size_t)2 * PB * PN * 4);
    __bf16* K  = Q + (size_t)PB * PN * PD;
    __bf16* V2 = K + (size_t)PB * PN * PD;

    {   // 1) QKV projection
        dim3 grid(PN / 64, (PD + PD + PC) / 16, PB);
        pam_proj<<<grid, 128, 0, stream>>>(x, Wq, bq, Wk, bk, Wv, bv, Q, K, V2);
    }
    {   // 2) softmax stats
        dim3 grid(PN / 128, PB);
        pam_stats<<<grid, 256, 0, stream>>>(Q, K, stats_m, stats_s);
    }
    {   // 3) attention output + residual
        dim3 grid(PN / 32, PB);
        pam_out<<<grid, 256, 0, stream>>>(x, Q, K, V2, stats_m, stats_s, gamma, out);
    }
}
